// cross_mha_add_66529043415420
// MI455X (gfx1250) — matrix operand order repaired and checked by reading, not compiled, not run on hardware
//
#include <hip/hip_runtime.h>

#define B_   2
#define SQ_  2048
#define SK_  2048
#define DM_  1024
#define H_   16
#define D_   64

typedef __attribute__((ext_vector_type(16))) __bf16 v16bf;
typedef __attribute__((ext_vector_type(8)))  __bf16 v8bf;
typedef __attribute__((ext_vector_type(8)))  float  v8f;

static __device__ __forceinline__ __bf16 f2bf(float x) { return (__bf16)x; }

static __device__ __forceinline__ v16bf cat16(v8bf lo, v8bf hi) {
  return __builtin_shufflevector(lo, hi, 0,1,2,3,4,5,6,7,8,9,10,11,12,13,14,15);
}

static __device__ __forceinline__ v8f wmma_bf16(v16bf a, v16bf b, v8f c) {
  // D = A(16x32 bf16) * B(32x16 bf16) + C(16x16 f32)
  return __builtin_amdgcn_wmma_f32_16x16x32_bf16(false, a, false, b, (short)0, c, false, false);
}

// Async global->LDS copy of 16 bytes per lane (512B per wave instruction),
// tracked by ASYNCcnt. lds_off = LDS byte offset, gaddr = 64-bit global addr.
static __device__ __forceinline__ void async_load_b128(unsigned lds_off, unsigned long long gaddr) {
  asm volatile("global_load_async_to_lds_b128 %0, %1, off"
               :: "v"(lds_off), "v"(gaddr) : "memory");
}
static __device__ __forceinline__ void wait_asynccnt0() {
  asm volatile("s_wait_asynccnt 0" ::: "memory");
}

// ---------------------------------------------------------------------------
// Kernel 1: per-head projection  Y[b,h,s,:] = X[b,s,:] @ W[h,:,:] + bias[h,:]
// X: [B,S,1024] f32, W: [H,1024,64] f32, bias: [H,64] f32, Y: [B,H,S,64] bf16
// Block: 256 threads (8 waves); wave w -> rows [tile*128 + 16w, +16), cols 0..63
// ---------------------------------------------------------------------------
__global__ __launch_bounds__(256) void qkv_proj_kernel(
    const float* __restrict__ X, const float* __restrict__ W,
    const float* __restrict__ bias, __bf16* __restrict__ Y, int S)
{
  __shared__ __attribute__((aligned(64))) __bf16 WtT[64][32];  // WtT[d][m_local]
  const int b = blockIdx.z, h = blockIdx.y;
  const int tid = threadIdx.x;
  const int wave = tid >> 5, lane = tid & 31;
  const int hl = lane >> 4, l16 = lane & 15;
  const int row0 = blockIdx.x * 128 + wave * 16;
  const float* Xb = X + (size_t)b * S * DM_;
  const float* Wh = W + (size_t)h * DM_ * D_;

  v8f acc[4] = {};

  const int s_ml = tid >> 3;        // 0..31 (k-local row of W slab)
  const int s_d  = (tid & 7) * 8;   // 0..56 (d column base)

  for (int kk = 0; kk < DM_; kk += 32) {
    // stage W[kk..kk+32, 0..64) -> LDS transposed bf16
    {
      const float* src = Wh + (size_t)(kk + s_ml) * D_ + s_d;
      float4 f0 = *(const float4*)src;
      float4 f1 = *(const float4*)(src + 4);
      WtT[s_d+0][s_ml] = f2bf(f0.x); WtT[s_d+1][s_ml] = f2bf(f0.y);
      WtT[s_d+2][s_ml] = f2bf(f0.z); WtT[s_d+3][s_ml] = f2bf(f0.w);
      WtT[s_d+4][s_ml] = f2bf(f1.x); WtT[s_d+5][s_ml] = f2bf(f1.y);
      WtT[s_d+6][s_ml] = f2bf(f1.z); WtT[s_d+7][s_ml] = f2bf(f1.w);
    }
    if (kk + 32 < DM_) {
      __builtin_prefetch(Wh + (size_t)(kk + 32 + s_ml) * D_ + s_d, 0, 0);
    }
    __syncthreads();
    // A fragment: lane holds row M=l16; e -> K = (hl?16:0)+e (contiguous)
    const float* ar = Xb + (size_t)(row0 + l16) * DM_ + kk + (hl ? 16 : 0);
    float4 a0 = *(const float4*)ar;
    float4 a1 = *(const float4*)(ar + 4);
    float4 a2 = *(const float4*)(ar + 8);
    float4 a3 = *(const float4*)(ar + 12);
    v16bf a;
    a[0]=f2bf(a0.x);  a[1]=f2bf(a0.y);  a[2]=f2bf(a0.z);  a[3]=f2bf(a0.w);
    a[4]=f2bf(a1.x);  a[5]=f2bf(a1.y);  a[6]=f2bf(a1.z);  a[7]=f2bf(a1.w);
    a[8]=f2bf(a2.x);  a[9]=f2bf(a2.y);  a[10]=f2bf(a2.z); a[11]=f2bf(a2.w);
    a[12]=f2bf(a3.x); a[13]=f2bf(a3.y); a[14]=f2bf(a3.z); a[15]=f2bf(a3.w);
#pragma unroll
    for (int nb = 0; nb < 4; ++nb) {
      // B frag: lane holds col N=16nb+l16; e -> K = (hl?16:0)+e (contiguous)
      v16bf bfrag = *(const v16bf*)(&WtT[16*nb + l16][hl ? 16 : 0]);
      acc[nb] = wmma_bf16(a, bfrag, acc[nb]);
    }
    __syncthreads();
  }
  __bf16* Yb = Y + (((size_t)b * H_ + h) * S + row0) * D_;
#pragma unroll
  for (int nb = 0; nb < 4; ++nb) {
    float bb = bias[h * D_ + 16*nb + l16];
#pragma unroll
    for (int r = 0; r < 8; ++r) {
      int m = r + (hl ? 8 : 0);  // C/D layout: VGPR r -> M = r + half*8
      Yb[(size_t)m * D_ + 16*nb + l16] = f2bf(acc[nb][r] + bb);
    }
  }
}

// ---------------------------------------------------------------------------
// Kernel 2: flash attention per (b,h).  Q,K,V: [B,H,S,64] bf16.
// O (concat-heads): [B,SQ,H*64] bf16.  Block = 128 q-rows (8 waves x 16).
// K tile staged via ASYNC global->LDS DMA; V tile transposed through VGPRs.
// ---------------------------------------------------------------------------
__global__ __launch_bounds__(256) void attn_kernel(
    const __bf16* __restrict__ Q, const __bf16* __restrict__ K,
    const __bf16* __restrict__ V, __bf16* __restrict__ O)
{
  __shared__ __attribute__((aligned(64))) __bf16 Kt[32][64];      // Kt[j_local][d]
  __shared__ __attribute__((aligned(64))) __bf16 Vt[64][32];      // Vt[d][j_local]
  __shared__ __attribute__((aligned(64))) __bf16 Pst[8][16][32];  // per-wave P scratch

  const int b = blockIdx.z, h = blockIdx.y;
  const int tid = threadIdx.x;
  const int wave = tid >> 5, lane = tid & 31;
  const int hl = lane >> 4, l16 = lane & 15;
  const int q0 = blockIdx.x * 128 + wave * 16;

  const __bf16* Qbh = Q + ((size_t)b * H_ + h) * SQ_ * D_;
  const __bf16* Kbh = K + ((size_t)b * H_ + h) * SK_ * D_;
  const __bf16* Vbh = V + ((size_t)b * H_ + h) * SK_ * D_;

  // LDS byte offset for this thread's 16B slice of the K tile
  const unsigned kt_off = (unsigned)(uintptr_t)(&Kt[0][0]) + (unsigned)tid * 16u;

  // Load Q A-fragments once, pre-scaled by 1/sqrt(64) = 2^-3 (exact in bf16)
  const float scale = 0.125f;
  v16bf qa[2];
#pragma unroll
  for (int c = 0; c < 2; ++c) {
    const __bf16* qr = Qbh + (size_t)(q0 + l16) * D_ + 32*c + (hl ? 16 : 0);
    v8bf lo = *(const v8bf*)qr;
    v8bf hi = *(const v8bf*)(qr + 8);
#pragma unroll
    for (int e = 0; e < 8; ++e) {
      qa[c][e]   = f2bf((float)lo[e] * scale);
      qa[c][8+e] = f2bf((float)hi[e] * scale);
    }
  }

  float m_i[8], l_i[8];
#pragma unroll
  for (int r = 0; r < 8; ++r) { m_i[r] = -3.0e38f; l_i[r] = 0.0f; }
  v8f accO[4] = {};

  const int s_jl = tid >> 3;       // 0..31
  const int s_d  = (tid & 7) * 8;  // 0..56

  for (int j0 = 0; j0 < SK_; j0 += 32) {
    // ---- stage K tile via async DMA (4KB: 16B/lane x 256 threads) ----
    async_load_b128(kt_off,
        (unsigned long long)(uintptr_t)(Kbh + (size_t)j0 * D_) + (unsigned)tid * 16u);
    // ---- stage V tile transposed via VGPRs ----
    {
      v8bf vv = *(const v8bf*)(Vbh + (size_t)(j0 + s_jl) * D_ + s_d);
#pragma unroll
      for (int e = 0; e < 8; ++e) Vt[s_d + e][s_jl] = vv[e];
    }
    wait_asynccnt0();
    __syncthreads();

    // ---- scores: S[16 x 32] = Q(16x64) . K^T ----
    v8f s[2];
#pragma unroll
    for (int t = 0; t < 2; ++t) {
      v8f c = {};
#pragma unroll
      for (int ch = 0; ch < 2; ++ch) {
        // B frag: lane -> key n = j0+16t+l16; e -> d = 32ch+(hl?16:0)+e
        v16bf kb = *(const v16bf*)(&Kt[16*t + l16][32*ch + (hl ? 16 : 0)]);
        c = wmma_bf16(qa[ch], kb, c);
      }
      s[t] = c;
    }

    // ---- online softmax (rows live in VGPR index r, spread over 16 lanes) ----
    float p0[8], p1[8];
#pragma unroll
    for (int r = 0; r < 8; ++r) {
      float v = fmaxf(s[0][r], s[1][r]);
      v = fmaxf(v, __shfl_xor(v, 1, 32));
      v = fmaxf(v, __shfl_xor(v, 2, 32));
      v = fmaxf(v, __shfl_xor(v, 4, 32));
      v = fmaxf(v, __shfl_xor(v, 8, 32));
      float nm  = fmaxf(m_i[r], v);
      float fac = __expf(m_i[r] - nm);
      m_i[r] = nm;
      p0[r] = __expf(s[0][r] - nm);
      p1[r] = __expf(s[1][r] - nm);
      float rs = p0[r] + p1[r];
      rs += __shfl_xor(rs, 1, 32);
      rs += __shfl_xor(rs, 2, 32);
      rs += __shfl_xor(rs, 4, 32);
      rs += __shfl_xor(rs, 8, 32);
      l_i[r] = l_i[r] * fac + rs;
#pragma unroll
      for (int nb = 0; nb < 4; ++nb) accO[nb][r] *= fac;
    }

    // ---- transpose P: C-layout -> A-layout via per-wave LDS scratch ----
#pragma unroll
    for (int r = 0; r < 8; ++r) {
      int m = r + (hl ? 8 : 0);
      Pst[wave][m][l16]      = f2bf(p0[r]);
      Pst[wave][m][16 + l16] = f2bf(p1[r]);
    }
    __builtin_amdgcn_wave_barrier();
    v16bf pa;
    {
      const __bf16* pr = &Pst[wave][l16][hl ? 16 : 0];
      v8bf lo = *(const v8bf*)pr;
      v8bf hi = *(const v8bf*)(pr + 8);
      pa = cat16(lo, hi);
    }

    // ---- O += P(16x32) . V(32x64) ----
#pragma unroll
    for (int nb = 0; nb < 4; ++nb) {
      // B frag: lane -> col d = 16nb+l16; e -> key = (hl?16:0)+e
      v16bf vb = *(const v16bf*)(&Vt[16*nb + l16][hl ? 16 : 0]);
      accO[nb] = wmma_bf16(pa, vb, accO[nb]);
    }
    __syncthreads();
  }

  // ---- epilogue: normalize and write concat-head output ----
  __bf16* Ob = O + ((size_t)b * SQ_ + q0) * (H_ * D_) + h * D_;
#pragma unroll
  for (int r = 0; r < 8; ++r) {
    float inv = 1.0f / l_i[r];
    int m = r + (hl ? 8 : 0);
#pragma unroll
    for (int nb = 0; nb < 4; ++nb)
      Ob[(size_t)m * (H_ * D_) + 16*nb + l16] = f2bf(accO[nb][r] * inv);
  }
}

// ---------------------------------------------------------------------------
// Kernel 3: out[s,i] = sum_j Oc[s,j] * Wo[i,j] + bo[i]
// Oc: [4096,1024] bf16, Wo: [1024,1024] f32, out: [4096,1024] f32
// ---------------------------------------------------------------------------
__global__ __launch_bounds__(256) void oproj_kernel(
    const __bf16* __restrict__ Oc, const float* __restrict__ Wo,
    const float* __restrict__ bo, float* __restrict__ out)
{
  __shared__ __attribute__((aligned(64))) __bf16 Wt[64][32];  // Wt[i_local][j_local]
  const int tid = threadIdx.x;
  const int wave = tid >> 5, lane = tid & 31;
  const int hl = lane >> 4, l16 = lane & 15;
  const int row0 = blockIdx.x * 128 + wave * 16;
  const int n0 = blockIdx.y * 64;

  v8f acc[4] = {};
  const int s_il = tid >> 2;        // 0..63
  const int s_jl = (tid & 3) * 8;   // 0..24

  for (int kk = 0; kk < DM_; kk += 32) {
    // stage Wo[n0..n0+64, kk..kk+32) -> LDS bf16 (contiguous in j)
    {
      const float* src = Wo + (size_t)(n0 + s_il) * DM_ + kk + s_jl;
      float4 f0 = *(const float4*)src;
      float4 f1 = *(const float4*)(src + 4);
      v8bf w;
      w[0]=f2bf(f0.x); w[1]=f2bf(f0.y); w[2]=f2bf(f0.z); w[3]=f2bf(f0.w);
      w[4]=f2bf(f1.x); w[5]=f2bf(f1.y); w[6]=f2bf(f1.z); w[7]=f2bf(f1.w);
      *(v8bf*)&Wt[s_il][s_jl] = w;
    }
    if (kk + 32 < DM_) {
      __builtin_prefetch(Wo + (size_t)(n0 + s_il) * DM_ + kk + 32 + s_jl, 0, 0);
    }
    __syncthreads();
    // A frag from Oc (bf16, contiguous groups of 8)
    const __bf16* ar = Oc + (size_t)(row0 + l16) * DM_ + kk + (hl ? 16 : 0);
    v8bf lo = *(const v8bf*)ar;
    v8bf hi = *(const v8bf*)(ar + 8);
    v16bf a = cat16(lo, hi);
#pragma unroll
    for (int nb = 0; nb < 4; ++nb) {
      // B frag: lane -> i = n0+16nb+l16; e -> j = kk+(hl?16:0)+e
      v16bf bfrag = *(const v16bf*)(&Wt[16*nb + l16][hl ? 16 : 0]);
      acc[nb] = wmma_bf16(a, bfrag, acc[nb]);
    }
    __syncthreads();
  }
#pragma unroll
  for (int nb = 0; nb < 4; ++nb) {
    float bb = bo[n0 + 16*nb + l16];
#pragma unroll
    for (int r = 0; r < 8; ++r) {
      int m = r + (hl ? 8 : 0);
      out[(size_t)(row0 + m) * DM_ + n0 + 16*nb + l16] = acc[nb][r] + bb;
    }
  }
}

// ---------------------------------------------------------------------------
extern "C" void kernel_launch(void* const* d_in, const int* in_sizes, int n_in,
                              void* d_out, int out_size, void* d_ws, size_t ws_size,
                              hipStream_t stream) {
  const float* x  = (const float*)d_in[0];   // [B,SK,1024]
  const float* x2 = (const float*)d_in[1];   // [B,SQ,1024]
  const float* Wq = (const float*)d_in[2];   // [H,1024,64]
  const float* bq = (const float*)d_in[3];   // [H,64]
  const float* Wk = (const float*)d_in[4];
  const float* bk = (const float*)d_in[5];
  const float* Wv = (const float*)d_in[6];
  const float* bv = (const float*)d_in[7];
  const float* Wo = (const float*)d_in[8];   // [1024,1024]
  const float* bo = (const float*)d_in[9];   // [1024]
  float* outp = (float*)d_out;               // [B,SQ,1024] f32

  const size_t nper = (size_t)B_ * H_ * SQ_ * D_;   // 4,194,304 elems
  __bf16* Qbf = (__bf16*)d_ws;
  __bf16* Kbf = Qbf + nper;
  __bf16* Vbf = Kbf + nper;
  __bf16* Obf = Vbf + nper;                          // [B,SQ,1024] bf16

  dim3 blk(256);
  dim3 gproj(SQ_ / 128, H_, B_);    // (16,16,2)
  qkv_proj_kernel<<<gproj, blk, 0, stream>>>(x2, Wq, bq, Qbf, SQ_);
  qkv_proj_kernel<<<gproj, blk, 0, stream>>>(x,  Wk, bk, Kbf, SK_);
  qkv_proj_kernel<<<gproj, blk, 0, stream>>>(x,  Wv, bv, Vbf, SK_);

  attn_kernel<<<dim3(SQ_ / 128, H_, B_), blk, 0, stream>>>(Qbf, Kbf, Vbf, Obf);

  oproj_kernel<<<dim3((B_ * SQ_) / 128, DM_ / 64), blk, 0, stream>>>(Obf, Wo, bo, outp);
}
